// RWKVLayer_87892210745633
// MI455X (gfx1250) — matrix operand order repaired and checked by reading, not compiled, not run on hardware
//
#include <hip/hip_runtime.h>
#include <hip/hip_bf16.h>
#include <cstdint>
#include <cstddef>

// Problem shape (fixed by reference)
#define B_DIM 8
#define T_DIM 2048
#define C_DIM 1024
#define F_DIM 4096
#define M_DIM (B_DIM * T_DIM)   // 16384 rows

typedef __bf16 bf16t;
typedef bf16t v16bf __attribute__((ext_vector_type(16)));
typedef float v8f __attribute__((ext_vector_type(8)));

union FragAB { int4 q[2]; v16bf v; };

__device__ __forceinline__ unsigned short f2bf(float f) {
  unsigned int u = __float_as_uint(f);
  u += 0x7fffu + ((u >> 16) & 1u);       // round-to-nearest-even
  return (unsigned short)(u >> 16);
}

enum { EPI_NONE = 0, EPI_ADD = 1, EPI_RELU2H = 2, EPI_GATED = 3 };

// ---------------------------------------------------------------------------
// fp32 [R, Ccols] -> bf16 transposed [Ccols, R]   (weight prep)
// ---------------------------------------------------------------------------
__global__ __launch_bounds__(256)
void convT_kernel(const float* __restrict__ in, unsigned short* __restrict__ out,
                  int R, int Ccols) {
  __shared__ float tile[32][33];
  const int c0 = blockIdx.x * 32;
  const int r0 = blockIdx.y * 32;
  const int tx = threadIdx.x;   // 0..31
  const int ty = threadIdx.y;   // 0..7
#pragma unroll
  for (int i = 0; i < 32; i += 8)
    tile[ty + i][tx] = in[(size_t)(r0 + ty + i) * Ccols + c0 + tx];
  __syncthreads();
#pragma unroll
  for (int i = 0; i < 32; i += 8)
    out[(size_t)(c0 + ty + i) * R + r0 + tx] = f2bf(tile[tx][ty + i]);
}

// ---------------------------------------------------------------------------
// Fused LayerNorm + token-shift + lerp-mix -> up to 3 bf16 outputs
// one block per (b,t) row; 256 threads, 4 channels/thread (C=1024)
// ---------------------------------------------------------------------------
template <int NOUT>
__global__ __launch_bounds__(256)
void ln_shift_mix_kernel(const float* __restrict__ x,
                         const float* __restrict__ gw, const float* __restrict__ bw,
                         const float* __restrict__ mv0, const float* __restrict__ mv1,
                         const float* __restrict__ mv2,
                         unsigned short* __restrict__ o0, unsigned short* __restrict__ o1,
                         unsigned short* __restrict__ o2) {
  const int C = C_DIM;
  const int bt = blockIdx.x;
  const int t = bt & (T_DIM - 1);
  const int tid = threadIdx.x;

  const float* rowT = x + (size_t)bt * C;
  float4 xt = *(const float4*)(rowT + tid * 4);
  float4 xp = make_float4(0.f, 0.f, 0.f, 0.f);
  if (t > 0) xp = *(const float4*)(rowT - C + tid * 4);

  __shared__ float4 red[256];
  float4 part;
  part.x = xt.x + xt.y + xt.z + xt.w;
  part.y = xt.x * xt.x + xt.y * xt.y + xt.z * xt.z + xt.w * xt.w;
  part.z = xp.x + xp.y + xp.z + xp.w;
  part.w = xp.x * xp.x + xp.y * xp.y + xp.z * xp.z + xp.w * xp.w;
  red[tid] = part;
  __syncthreads();
  for (int s = 128; s > 0; s >>= 1) {
    if (tid < s) {
      float4 a = red[tid], c = red[tid + s];
      a.x += c.x; a.y += c.y; a.z += c.z; a.w += c.w;
      red[tid] = a;
    }
    __syncthreads();
  }
  const float4 tot = red[0];
  const float invC = 1.0f / (float)C;
  const float mT = tot.x * invC;
  const float rT = rsqrtf(tot.y * invC - mT * mT + 1e-5f);
  const float mP = tot.z * invC;
  const float rP = rsqrtf(tot.w * invC - mP * mP + 1e-5f);

  const float* xtp = (const float*)&xt;
  const float* xpp = (const float*)&xp;
#pragma unroll
  for (int e = 0; e < 4; e++) {
    const int c = tid * 4 + e;
    const float g = gw[c], bb = bw[c];
    const float xa = (xtp[e] - mT) * rT * g + bb;
    const float xs = (t > 0) ? ((xpp[e] - mP) * rP * g + bb) : 0.f;
    const size_t off = (size_t)bt * C + c;
    { const float m = mv0[c]; o0[off] = f2bf(m * xa + (1.f - m) * xs); }
    if (NOUT > 1) { const float m = mv1[c]; o1[off] = f2bf(m * xa + (1.f - m) * xs); }
    if (NOUT > 2) { const float m = mv2[c]; o2[off] = f2bf(m * xa + (1.f - m) * xs); }
  }
}

// ---------------------------------------------------------------------------
// bf16 WMMA GEMM: out[M,N] = A[M,K] @ Bt[N,K]^T   (Bt = transposed weight)
// one wave -> 64x64 output tile (4x4 wmma); 8 waves / block as 2(m) x 4(n)
// -> block tile 128 x 256. Halves weight-side L2 traffic vs 32-row tiles.
// ---------------------------------------------------------------------------
template <int EPI>
__global__ __launch_bounds__(256)
void gemm_bf16_wmma(const unsigned short* __restrict__ A,
                    const unsigned short* __restrict__ Bt,
                    const float* __restrict__ base,
                    const float* __restrict__ gate,
                    float* __restrict__ outF,
                    unsigned short* __restrict__ outH,
                    int N, int K) {
  const int wave = threadIdx.x >> 5;
  const int lane = threadIdx.x & 31;
  const int nBlkN = N >> 8;                 // block tiles along N (256 wide)
  const int bm = blockIdx.x / nBlkN;
  const int bn = blockIdx.x - bm * nBlkN;
  const int m0 = bm * 128 + (wave >> 2) * 64;
  const int n0 = bn * 256 + (wave & 3) * 64;
  const int hf = lane >> 4;    // half-wave id
  const int lr = lane & 15;

  // A fragment (16x32 bf16): per-lane two b128 chunks at k+8*hf and k+8*hf+16
  const unsigned short* ap = A + (size_t)(m0 + lr) * K + hf * 8;
  // B fragment (32x16): per-lane two b128 chunks at k+8*hf and k+8*hf+16 on row n of Bt
  const unsigned short* bp = Bt + (size_t)(n0 + lr) * K + hf * 8;
  const size_t stride16 = (size_t)16 * K;   // 16 rows

  v8f acc[4][4] = {};

#pragma unroll 1
  for (int kb = 0; kb < K; kb += 32) {
    FragAB a[4], b[4];
#pragma unroll
    for (int i = 0; i < 4; i++) {
      a[i].q[0] = *(const int4*)(ap + i * stride16 + kb);
      a[i].q[1] = *(const int4*)(ap + i * stride16 + kb + 16);
    }
#pragma unroll
    for (int j = 0; j < 4; j++) {
      b[j].q[0] = *(const int4*)(bp + j * stride16 + kb);
      b[j].q[1] = *(const int4*)(bp + j * stride16 + kb + 16);
    }
#pragma unroll
    for (int i = 0; i < 4; i++) {
#pragma unroll
      for (int j = 0; j < 4; j++) {
        acc[i][j] = __builtin_amdgcn_wmma_f32_16x16x32_bf16(
            false, a[i].v, false, b[j].v, (short)0, acc[i][j], false, false);
      }
    }
  }

  // Epilogue. D layout: vgpr r -> row m0+16*i+8*hf+r, col n0+16*j+lr
#pragma unroll
  for (int i = 0; i < 4; i++) {
#pragma unroll
    for (int j = 0; j < 4; j++) {
#pragma unroll
      for (int r = 0; r < 8; r++) {
        const int row = m0 + 16 * i + 8 * hf + r;
        const int col = n0 + 16 * j + lr;
        const size_t off = (size_t)row * N + col;
        const float vacc = acc[i][j][r];
        if (EPI == EPI_NONE) {
          outF[off] = vacc;
        } else if (EPI == EPI_ADD) {
          outF[off] = base[off] + vacc;
        } else if (EPI == EPI_RELU2H) {
          const float tr = vacc > 0.f ? vacc : 0.f;
          outH[off] = f2bf(tr * tr);
        } else {  // EPI_GATED: base + sigmoid(gate) * acc
          const float g = gate[off];
          const float s = 1.f / (1.f + __expf(-g));
          outF[off] = base[off] + s * vacc;
        }
      }
    }
  }
}

// ---------------------------------------------------------------------------
// WKV recurrence: segmented associative scan on the stable (a,b,p) state.
// Block = 16 channels x 16 segments (256 threads). Output = sigmoid(r)*wkv
// written as bf16 (A-matrix for the Wo GEMM).
// ---------------------------------------------------------------------------
__global__ __launch_bounds__(256)
void wkv_kernel(const float* __restrict__ Kin, const float* __restrict__ Vin,
                const float* __restrict__ Rin,
                const float* __restrict__ time_decay, const float* __restrict__ time_first,
                unsigned short* __restrict__ OUT) {
  const int C = C_DIM, T = T_DIM;
  const int L = T / 16;  // 128 steps per segment
  const int nblkC = C / 16;
  const int b = blockIdx.x / nblkC;
  const int c0 = (blockIdx.x % nblkC) * 16;
  const int tid = threadIdx.x;
  const int cl = tid & 15;
  const int seg = tid >> 4;
  const int c = c0 + cl;

  const float w = -__expf(time_decay[c]);
  const float u = time_first[c];
  const size_t baseIdx = (size_t)b * T * C + c;

  // Pass 1: per-segment aggregate from identity state
  float aa = 0.f, bb = 0.f, pp = -1e38f;
  size_t idx = baseIdx + (size_t)seg * L * C;
  for (int t = 0; t < L; t++) {
    const float kt = Kin[idx], vt = Vin[idx];
    const float ww = pp + w;
    const float p2 = fmaxf(ww, kt);
    const float e1 = __expf(ww - p2);
    const float e2 = __expf(kt - p2);
    aa = e1 * aa + e2 * vt;
    bb = e1 * bb + e2;
    pp = p2;
    idx += C;
  }

  __shared__ float sA[16][16], sB[16][16], sP[16][16];
  __shared__ float pA[16][16], pB[16][16], pP[16][16];
  sA[seg][cl] = aa; sB[seg][cl] = bb; sP[seg][cl] = pp;
  __syncthreads();

  // Pass 2: exclusive prefix over segments (16 threads, one per channel)
  if (tid < 16) {
    float a2 = 0.f, b2 = 0.f, p2s = -1e38f;
    const float decL = (float)L * w;
    for (int s = 0; s < 16; s++) {
      pA[s][tid] = a2; pB[s][tid] = b2; pP[s][tid] = p2s;
      const float pd = p2s + decL;
      const float pa = sP[s][tid];
      const float pm = fmaxf(pd, pa);
      const float e1 = __expf(pd - pm);
      const float e2 = __expf(pa - pm);
      a2 = e1 * a2 + e2 * sA[s][tid];
      b2 = e1 * b2 + e2 * sB[s][tid];
      p2s = pm;
    }
  }
  __syncthreads();

  // Pass 3: replay segment with incoming prefix, emit sigmoid(r)*wkv
  aa = pA[seg][cl]; bb = pB[seg][cl]; pp = pP[seg][cl];
  idx = baseIdx + (size_t)seg * L * C;
  for (int t = 0; t < L; t++) {
    const float kt = Kin[idx], vt = Vin[idx];
    float ww = u + kt;
    float p = fmaxf(pp, ww);
    float e1 = __expf(pp - p);
    float e2 = __expf(ww - p);
    const float wkv = (e1 * aa + e2 * vt) / (e1 * bb + e2);
    const float rp = Rin[idx];
    const float s = 1.f / (1.f + __expf(-rp));
    OUT[idx] = f2bf(s * wkv);
    ww = pp + w;
    p = fmaxf(ww, kt);
    e1 = __expf(ww - p);
    e2 = __expf(kt - p);
    aa = e1 * aa + e2 * vt;
    bb = e1 * bb + e2;
    pp = p;
    idx += C;
  }
}

// ---------------------------------------------------------------------------
extern "C" void kernel_launch(void* const* d_in, const int* in_sizes, int n_in,
                              void* d_out, int out_size, void* d_ws, size_t ws_size,
                              hipStream_t stream) {
  const int C = C_DIM, F = F_DIM, M = M_DIM;

  const float* x    = (const float*)d_in[0];
  const float* ln0g = (const float*)d_in[1];
  const float* ln0b = (const float*)d_in[2];
  const float* ln1g = (const float*)d_in[3];
  const float* ln1b = (const float*)d_in[4];
  const float* tdec = (const float*)d_in[5];
  const float* tfir = (const float*)d_in[6];
  const float* tmk  = (const float*)d_in[7];
  const float* tmv  = (const float*)d_in[8];
  const float* tmr  = (const float*)d_in[9];
  const float* Wk   = (const float*)d_in[10];
  const float* Wv   = (const float*)d_in[11];
  const float* Wr   = (const float*)d_in[12];
  const float* Wo   = (const float*)d_in[13];
  const float* cmk  = (const float*)d_in[14];
  const float* cmr  = (const float*)d_in[15];
  const float* Wck  = (const float*)d_in[16];
  const float* Wcv  = (const float*)d_in[17];
  const float* Wcr  = (const float*)d_in[18];
  float* out = (float*)d_out;

  // Carve workspace
  char* p = (char*)d_ws;
  auto carve = [&](size_t bytes) -> void* {
    void* r = (void*)p;
    p += (bytes + 255) & ~(size_t)255;
    return r;
  };
  unsigned short* wkT  = (unsigned short*)carve((size_t)C * C * 2);
  unsigned short* wvT  = (unsigned short*)carve((size_t)C * C * 2);
  unsigned short* wrT  = (unsigned short*)carve((size_t)C * C * 2);
  unsigned short* woT  = (unsigned short*)carve((size_t)C * C * 2);
  unsigned short* wcrT = (unsigned short*)carve((size_t)C * C * 2);
  unsigned short* wckT = (unsigned short*)carve((size_t)F * C * 2);  // [F,C]
  unsigned short* wcvT = (unsigned short*)carve((size_t)C * F * 2);  // [C,F]
  unsigned short* xA0  = (unsigned short*)carve((size_t)M * C * 2);  // xk -> rwkv
  unsigned short* xA1  = (unsigned short*)carve((size_t)M * C * 2);  // xv -> xck
  unsigned short* xA2  = (unsigned short*)carve((size_t)M * C * 2);  // xr -> xcr
  float* fK  = (float*)carve((size_t)M * C * 4);                     // k  -> rr_pre
  float* fV  = (float*)carve((size_t)M * C * 4);                     // v
  float* fR  = (float*)carve((size_t)M * C * 4);                     // r_pre
  float* fX1 = (float*)carve((size_t)M * C * 4);                     // x after attention
  unsigned short* kact = (unsigned short*)carve((size_t)M * F * 2);  // relu(kk)^2

  const dim3 tb(32, 8);
  // Weight conversion (fp32 [R,Ccols] -> bf16 [Ccols,R])
  convT_kernel<<<dim3(C / 32, C / 32), tb, 0, stream>>>(Wk,  wkT,  C, C);
  convT_kernel<<<dim3(C / 32, C / 32), tb, 0, stream>>>(Wv,  wvT,  C, C);
  convT_kernel<<<dim3(C / 32, C / 32), tb, 0, stream>>>(Wr,  wrT,  C, C);
  convT_kernel<<<dim3(C / 32, C / 32), tb, 0, stream>>>(Wo,  woT,  C, C);
  convT_kernel<<<dim3(C / 32, C / 32), tb, 0, stream>>>(Wcr, wcrT, C, C);
  convT_kernel<<<dim3(F / 32, C / 32), tb, 0, stream>>>(Wck, wckT, C, F);
  convT_kernel<<<dim3(C / 32, F / 32), tb, 0, stream>>>(Wcv, wcvT, F, C);

  // Time-mix: LN0 + shift + lerp -> bf16 xk/xv/xr
  ln_shift_mix_kernel<3><<<M, 256, 0, stream>>>(x, ln0g, ln0b, tmk, tmv, tmr,
                                                xA0, xA1, xA2);

  const int blkCC = (M / 128) * (C / 256);  // 512 blocks
  const int blkCF = (M / 128) * (F / 256);  // 2048 blocks

  // k, v, r projections (f32 outputs)
  gemm_bf16_wmma<EPI_NONE><<<blkCC, 256, 0, stream>>>(xA0, wkT, nullptr, nullptr, fK, nullptr, C, C);
  gemm_bf16_wmma<EPI_NONE><<<blkCC, 256, 0, stream>>>(xA1, wvT, nullptr, nullptr, fV, nullptr, C, C);
  gemm_bf16_wmma<EPI_NONE><<<blkCC, 256, 0, stream>>>(xA2, wrT, nullptr, nullptr, fR, nullptr, C, C);

  // WKV segmented scan; emits sigmoid(r)*wkv as bf16 into xA0
  wkv_kernel<<<B_DIM * (C / 16), 256, 0, stream>>>(fK, fV, fR, tdec, tfir, xA0);

  // x1 = x + rwkv @ Wo
  gemm_bf16_wmma<EPI_ADD><<<blkCC, 256, 0, stream>>>(xA0, woT, x, nullptr, fX1, nullptr, C, C);

  // Channel-mix: LN1 + shift + lerp -> bf16 xck/xcr
  ln_shift_mix_kernel<2><<<M, 256, 0, stream>>>(fX1, ln1g, ln1b, cmk, cmr, nullptr,
                                                xA1, xA2, nullptr);

  // kk = xck @ Wck ; relu^2 -> bf16
  gemm_bf16_wmma<EPI_RELU2H><<<blkCF, 256, 0, stream>>>(xA1, wckT, nullptr, nullptr, nullptr, kact, F, C);
  // rr_pre = xcr @ Wcr (reuse fK)
  gemm_bf16_wmma<EPI_NONE><<<blkCC, 256, 0, stream>>>(xA2, wcrT, nullptr, nullptr, fK, nullptr, C, C);
  // out = x1 + sigmoid(rr_pre) * (kact @ Wcv)
  gemm_bf16_wmma<EPI_GATED><<<blkCC, 256, 0, stream>>>(kact, wcvT, fX1, fK, out, nullptr, C, F);

  (void)in_sizes; (void)n_in; (void)out_size; (void)ws_size;
}
